// EnsembleMember_5686536700165
// MI455X (gfx1250) — matrix operand order repaired and checked by reading, not compiled, not run on hardware
//
#include <hip/hip_runtime.h>
#include <hip/hip_bf16.h>

// ---------------------------------------------------------------------------
// sLSTM x2 + FC head for MI455X (gfx1250), wave32 + WMMA f32_16x16x32_f16.
//
// Sizes: B=128, T=1024, D=3, H=512 (4H=2048), OUT=26.
//
// Latency-bound serial scan (T=1024) -> persistent kernel, weights in LDS,
// h double-buffered in global/L2, one grid barrier per step, recurrent
// state (c,n,m) held in VGPRs in WMMA C-layout for the whole scan.
// Inner loop is software-pipelined at depth 2: loads for K-tile k+2 are
// issued before the 4 WMMAs of K-tile k, so every LDS/L2 access has two
// full WMMA groups of latency cover.
// ---------------------------------------------------------------------------

typedef __attribute__((ext_vector_type(16))) _Float16 v16h;
typedef __attribute__((ext_vector_type(8)))  float    v8f;

#define Bn   128
#define Tn   1024
#define Dn   3
#define Hn   512
#define G4   2048          // 4H
#define KT   16            // 512 / 32 K-tiles
#define NBLK 32            // scan grid: one block per 16 h-columns
#define GATE_HALVES (KT*32*16)   // 8192 fp16 per (gate-nTile) slice = 16 KB

union AFrag { uint4 u4[2]; v16h v; };

// ---------------------------------------------------------------------------
// Pack a (2048 x 512) f32 weight into fp16 WMMA-B fragments.
// dst[(((nTile*KT)+kt)*32 + lane)*16 + h]  with
//   n = nTile*16 + lane%16 ;  k = kt*32 + (lane/16)*16 + h
// => each lane's 16 K-values are one contiguous 32B v16h.
// ---------------------------------------------------------------------------
__global__ void pack_b(const float* __restrict__ src, _Float16* __restrict__ dst) {
    int idx = blockIdx.x * blockDim.x + threadIdx.x;           // 0 .. 2048*512-1
    if (idx >= G4 * Hn) return;
    int h     =  idx        & 15;
    int lane  = (idx >> 4)  & 31;
    int kt    = (idx >> 9)  & 15;
    int nTile =  idx >> 13;
    int n = nTile * 16 + (lane & 15);
    int k = kt * 32 + (lane >> 4) * 16 + h;
    dst[idx] = (_Float16)src[n * Hn + k];
}

// ---------------------------------------------------------------------------
// Sense-reversing grid barrier (global atomics + s_sleep backoff).
// ---------------------------------------------------------------------------
__device__ __forceinline__ void grid_barrier(unsigned* cnt, unsigned* gen, unsigned nb) {
    __syncthreads();
    if (threadIdx.x == 0) {
        __threadfence();
        volatile unsigned* vgen = gen;
        unsigned g = *vgen;
        if (atomicAdd(cnt, 1u) == nb - 1u) {
            atomicExch(cnt, 0u);
            __threadfence();
            atomicAdd(gen, 1u);
        } else {
            while (*vgen == g) { __builtin_amdgcn_s_sleep(2); }
        }
        __threadfence();
    }
    __syncthreads();
}

// ---------------------------------------------------------------------------
// Persistent sLSTM scan. Block `blk` owns h-columns [blk*16, blk*16+16);
// wave `w` owns batch rows [w*16, w*16+16). Each wave accumulates the 4 gate
// tiles (i,f,z,o) for its 16x16 patch; c/n/m live in VGPRs across all steps.
// ---------------------------------------------------------------------------
template <int LAYER>
__global__ void __launch_bounds__(256, 1)
slstm_scan(const float*    __restrict__ x,        // (B,T,3)      [LAYER==0]
           const float*    __restrict__ W0,       // (2048,3)     [LAYER==0]
           const float*    __restrict__ bias,     // (2048,)
           const _Float16* __restrict__ packedR,  // recurrent weights, packed
           const _Float16* __restrict__ packedW1, // input weights, packed [LAYER==1]
           const _Float16* __restrict__ h1seq_in, // (T,B,H) fp16 [LAYER==1]
           _Float16*       __restrict__ h1seq_out,// (T,B,H) fp16 [LAYER==0]
           _Float16*       __restrict__ hbuf,     // (2,B,H) fp16 double buffer
           float*          __restrict__ h2last,   // (B,H) f32    [LAYER==1]
           unsigned* bar_cnt, unsigned* bar_gen)
{
    extern __shared__ _Float16 lds[];   // R slice (64KB) [+ W1 slice (64KB)]

    const int tid  = threadIdx.x;
    const int lane = tid & 31;
    const int wave = tid >> 5;
    const int blk  = blockIdx.x;

    // --- Stage packed weight slices into LDS (frag layout preserved) -------
    for (int g = 0; g < 4; ++g) {
        const uint4* srcR = (const uint4*)(packedR + (size_t)(g * 32 + blk) * GATE_HALVES);
        uint4*       dstR = (uint4*)(lds + g * GATE_HALVES);
        for (int i = tid; i < GATE_HALVES / 8; i += 256) dstR[i] = srcR[i];
        if (LAYER == 1) {
            const uint4* srcW = (const uint4*)(packedW1 + (size_t)(g * 32 + blk) * GATE_HALVES);
            uint4*       dstW = (uint4*)(lds + (4 + g) * GATE_HALVES);
            for (int i = tid; i < GATE_HALVES / 8; i += 256) dstW[i] = srcW[i];
        }
    }
    __syncthreads();

    // --- Per-lane loop-invariant constants ---------------------------------
    const int half = lane >> 4;               // A-frag K-group select
    const int rA   = wave * 16 + (lane & 15); // row this lane loads for A frags
    const int hc   = blk * 16 + (lane & 15);  // h column (C-layout N = lane%16)

    float bg[4];
    float w0g[4][3];
    #pragma unroll
    for (int g = 0; g < 4; ++g) {
        const int cgl = g * Hn + hc;
        bg[g] = bias[cgl];
        if (LAYER == 0) {
            w0g[g][0] = W0[cgl * 3 + 0];
            w0g[g][1] = W0[cgl * 3 + 1];
            w0g[g][2] = W0[cgl * 3 + 2];
        }
    }

    // --- Recurrent state in registers (WMMA C layout: VGPR e, this lane) ---
    float st_c[8], st_n[8], st_m[8];
    #pragma unroll
    for (int e = 0; e < 8; ++e) { st_c[e] = 0.f; st_n[e] = 0.f; st_m[e] = 0.f; }

    constexpr int KTOT = (LAYER == 1) ? 2 * KT : KT;   // layer1: R pass + W1 pass fused
    const size_t laneAOff = (size_t)rA * Hn + half * 16;
    const int    laneBOff = lane * 16;

    for (int t = 0; t < Tn; ++t) {
        const _Float16* hb  = hbuf + (size_t)(t & 1) * (Bn * Hn);
        const _Float16* h1t = (LAYER == 1) ? (h1seq_in + (size_t)t * (Bn * Hn)) : nullptr;

        // k2 in [0,KT): A = h_{t-1}, B = R ; k2 in [KT,2KT): A = h1_t, B = W1
        auto loadA = [&](int k2) -> AFrag {
            const _Float16* src = (LAYER == 1 && k2 >= KT) ? h1t : hb;
            const uint4* pa = (const uint4*)(src + laneAOff + (k2 & (KT - 1)) * 32);
            AFrag a;
            a.u4[0] = pa[0];      // K block +0
            a.u4[1] = pa[1];      // K block +8 halves
            return a;
        };
        auto loadB = [&](int k2, int g) -> v16h {
            const int slice = (k2 >= KT) ? (4 + g) : g;
            return *(const v16h*)(lds + (slice * KT + (k2 & (KT - 1))) * 512 + laneBOff);
        };

        v8f acc[4] = {};

        // ---- Depth-2 software-pipelined WMMA loop -------------------------
        AFrag aC = loadA(0);
        v16h  bC0 = loadB(0, 0), bC1 = loadB(0, 1), bC2 = loadB(0, 2), bC3 = loadB(0, 3);
        AFrag aN = loadA(1);
        v16h  bN0 = loadB(1, 0), bN1 = loadB(1, 1), bN2 = loadB(1, 2), bN3 = loadB(1, 3);
        #pragma unroll
        for (int k2 = 0; k2 < KTOT; ++k2) {
            AFrag aF = aN;
            v16h  bF0 = bN0, bF1 = bN1, bF2 = bN2, bF3 = bN3;
            if (k2 + 2 < KTOT) {        // issue loads two tiles ahead
                aF  = loadA(k2 + 2);
                bF0 = loadB(k2 + 2, 0);
                bF1 = loadB(k2 + 2, 1);
                bF2 = loadB(k2 + 2, 2);
                bF3 = loadB(k2 + 2, 3);
            }
            acc[0] = __builtin_amdgcn_wmma_f32_16x16x32_f16(false, aC.v, false, bC0, (short)0, acc[0], false, false);
            acc[1] = __builtin_amdgcn_wmma_f32_16x16x32_f16(false, aC.v, false, bC1, (short)0, acc[1], false, false);
            acc[2] = __builtin_amdgcn_wmma_f32_16x16x32_f16(false, aC.v, false, bC2, (short)0, acc[2], false, false);
            acc[3] = __builtin_amdgcn_wmma_f32_16x16x32_f16(false, aC.v, false, bC3, (short)0, acc[3], false, false);
            aC = aN;  bC0 = bN0; bC1 = bN1; bC2 = bN2; bC3 = bN3;
            aN = aF;  bN0 = bF0; bN1 = bF1; bN2 = bF2; bN3 = bF3;
        }

        // ---- Exponential gating + state update (fp32, per C-layout elem) --
        _Float16* hb_next = hbuf + (size_t)((t + 1) & 1) * (Bn * Hn);
        #pragma unroll
        for (int e = 0; e < 8; ++e) {
            const int r = wave * 16 + e + 8 * half;   // C layout: M = e (+8 for hi lanes)
            float gi = acc[0][e] + bg[0];
            float gf = acc[1][e] + bg[1];
            float gz = acc[2][e] + bg[2];
            float go = acc[3][e] + bg[3];
            if (LAYER == 0) {
                const float* xr = x + ((size_t)r * Tn + t) * Dn;
                const float x0 = xr[0], x1 = xr[1], x2 = xr[2];
                gi += w0g[0][0] * x0 + w0g[0][1] * x1 + w0g[0][2] * x2;
                gf += w0g[1][0] * x0 + w0g[1][1] * x1 + w0g[1][2] * x2;
                gz += w0g[2][0] * x0 + w0g[2][1] * x1 + w0g[2][2] * x2;
                go += w0g[3][0] * x0 + w0g[3][1] * x1 + w0g[3][2] * x2;
            }
            const float mn = fmaxf(gf + st_m[e], gi);
            const float ig = __expf(gi - mn);
            const float fg = __expf(gf + st_m[e] - mn);
            // tanh(z) = 1 - 2/(1+e^{2z})  (v_exp_f32 + v_rcp_f32, TRANS class)
            const float tz = 1.f - 2.f * __builtin_amdgcn_rcpf(1.f + __expf(2.f * gz));
            const float cn = fg * st_c[e] + ig * tz;
            const float nn = fg * st_n[e] + ig;
            const float sg = __builtin_amdgcn_rcpf(1.f + __expf(-go));
            const float hn = sg * cn * __builtin_amdgcn_rcpf(fmaxf(nn, 1e-6f));
            st_c[e] = cn; st_n[e] = nn; st_m[e] = mn;

            const _Float16 hh = (_Float16)hn;
            hb_next[(size_t)r * Hn + hc] = hh;
            if (LAYER == 0) h1seq_out[(size_t)t * (Bn * Hn) + (size_t)r * Hn + hc] = hh;
            if (LAYER == 1 && t == Tn - 1) h2last[(size_t)r * Hn + hc] = hn;
        }

        // ---- All blocks must see h_t before anyone reads it at t+1 --------
        grid_barrier(bar_cnt, bar_gen, NBLK);
    }
}

// ---------------------------------------------------------------------------
// FC head: out(128,26) = h2_last(128,512) @ fc_w^T + fc_b.  Tiny; plain VALU.
// ---------------------------------------------------------------------------
__global__ void fc_kernel(const float* __restrict__ h2,
                          const float* __restrict__ fcw,
                          const float* __restrict__ fcb,
                          float*       __restrict__ out)
{
    const int idx = blockIdx.x * blockDim.x + threadIdx.x;
    if (idx >= Bn * 26) return;
    const int b = idx / 26, o = idx % 26;
    const float* hp = h2  + (size_t)b * Hn;
    const float* wp = fcw + (size_t)o * Hn;
    float s = fcb[o];
    for (int k = 0; k < Hn; ++k) s += hp[k] * wp[k];
    out[idx] = s;
}

// ---------------------------------------------------------------------------
// Host glue (graph-capture safe: only kernel launches + hipMemsetAsync).
// ---------------------------------------------------------------------------
extern "C" void kernel_launch(void* const* d_in, const int* in_sizes, int n_in,
                              void* d_out, int out_size, void* d_ws, size_t ws_size,
                              hipStream_t stream)
{
    const float* x    = (const float*)d_in[0];
    const float* W0   = (const float*)d_in[1];
    const float* R0   = (const float*)d_in[2];
    const float* b0   = (const float*)d_in[3];
    const float* W1   = (const float*)d_in[4];
    const float* R1   = (const float*)d_in[5];
    const float* b1   = (const float*)d_in[6];
    const float* fcw  = (const float*)d_in[7];
    const float* fcb  = (const float*)d_in[8];
    float* out = (float*)d_out;

    // Workspace layout
    unsigned char* base = (unsigned char*)d_ws;
    const size_t PACK_BYTES = (size_t)G4 * Hn * sizeof(_Float16);  // 2 MB each
    _Float16* pR0    = (_Float16*)(base);
    _Float16* pW1    = (_Float16*)(base + PACK_BYTES);
    _Float16* pR1    = (_Float16*)(base + 2 * PACK_BYTES);
    _Float16* hbuf   = (_Float16*)(base + 3 * PACK_BYTES);                     // 256 KB
    const size_t HBUF_BYTES = (size_t)2 * Bn * Hn * sizeof(_Float16);
    float*    h2last = (float*)   (base + 3 * PACK_BYTES + HBUF_BYTES);        // 256 KB
    const size_t H2_BYTES = (size_t)Bn * Hn * sizeof(float);
    unsigned* bar    = (unsigned*)(base + 3 * PACK_BYTES + HBUF_BYTES + H2_BYTES);
    _Float16* h1seq  = (_Float16*)(base + 3 * PACK_BYTES + HBUF_BYTES + H2_BYTES + 256);
    // h1seq = 128 MB; total ws use ~141 MB.

    // Zero h0 double-buffer + h2last + barrier state (contiguous region).
    hipMemsetAsync(hbuf, 0, HBUF_BYTES + H2_BYTES + 256, stream);

    // Pack recurrent / input weights into WMMA-B fragment layout (fp16).
    const int packThreads = 256;
    const int packBlocks  = (G4 * Hn + packThreads - 1) / packThreads;
    pack_b<<<packBlocks, packThreads, 0, stream>>>(R0, pR0);
    pack_b<<<packBlocks, packThreads, 0, stream>>>(W1, pW1);
    pack_b<<<packBlocks, packThreads, 0, stream>>>(R1, pR1);

    // Layer 0 scan: LDS = R0 slice (64 KB).
    slstm_scan<0><<<NBLK, 256, 4 * GATE_HALVES * sizeof(_Float16), stream>>>(
        x, W0, b0, pR0, nullptr, nullptr, h1seq, hbuf, nullptr, bar, bar + 1);

    // Re-zero h double buffer for layer 1 (barrier counters self-reset).
    hipMemsetAsync(hbuf, 0, HBUF_BYTES, stream);

    // Layer 1 scan: LDS = R1 slice + W1 slice (128 KB); fused input proj.
    slstm_scan<1><<<NBLK, 256, 8 * GATE_HALVES * sizeof(_Float16), stream>>>(
        nullptr, nullptr, b1, pR1, pW1, h1seq, nullptr, hbuf, h2last, bar, bar + 1);

    // FC head.
    fc_kernel<<<(Bn * 26 + 255) / 256, 256, 0, stream>>>(h2last, fcw, fcb, out);
}
